// Qwen3_5SparseMoeBlock_10118942949429
// MI455X (gfx1250) — matrix operand order repaired and checked by reading, not compiled, not run on hardware
//
#include <hip/hip_runtime.h>
#include <hip/hip_bf16.h>

// ---------------------------------------------------------------------------
// Qwen3.5 sparse MoE block for gfx1250 (MI455X), wave32 + WMMA bf16 path.
// T=4096 tokens, H=2048 hidden, E=64 experts, top-K=8, I=512, IS=5632.
// Double-buffered LDS with async global->LDS staging for bf16 A tiles.
// ---------------------------------------------------------------------------

typedef __attribute__((ext_vector_type(16))) __bf16 v16bf;
typedef __attribute__((ext_vector_type(8)))  float  v8f;
typedef int v4i_ __attribute__((vector_size(16)));

constexpr int T_    = 4096;
constexpr int H_    = 2048;
constexpr int E_    = 64;
constexpr int TOPK  = 8;
constexpr int I_    = 512;
constexpr int IS_   = 5632;
constexpr int CAP   = T_ * TOPK + E_ * 64;   // 36864 padded dispatch rows
constexpr int MAXT  = CAP / 64;              // 576 max M-tiles (static grid)
constexpr int KSTR  = 40;                    // LDS row stride (bf16), 80B (16B aligned)

__device__ __forceinline__ float fast_sigmoid(float x) {
    return __builtin_amdgcn_rcpf(1.f + __expf(-x));   // v_rcp_f32, no div expansion
}
__device__ __forceinline__ float fast_silu(float x) { return x * fast_sigmoid(x); }

__device__ __forceinline__ v8f wmma_bf16(v16bf a, v16bf b, v8f c) {
    return __builtin_amdgcn_wmma_f32_16x16x32_bf16(false, a, false, b, (short)0, c, false, false);
}

// Async 16B global -> LDS copy (ASYNCcnt path); falls back to a plain copy.
__device__ __forceinline__ void async_copy16(const __bf16* gsrc, __bf16* ldst) {
#if __has_builtin(__builtin_amdgcn_global_load_async_to_lds_b128)
    __builtin_amdgcn_global_load_async_to_lds_b128(
        (__attribute__((address_space(1))) v4i_*)(void*)gsrc,
        (__attribute__((address_space(3))) v4i_*)(void*)ldst, 0, 0);
#else
    *(uint4*)ldst = *(const uint4*)gsrc;
#endif
}
__device__ __forceinline__ void wait_async() {
#if __has_builtin(__builtin_amdgcn_s_wait_asynccnt)
    __builtin_amdgcn_s_wait_asynccnt(0);
#endif
}

// A fragment (16x32 bf16, M x K): lane l<16 holds row M=l with K=0..7,16..23;
// lane l>=16 holds row M=l-16 with K=8..15,24..31.  lA is [m][k], stride KSTR.
__device__ __forceinline__ v16bf load_a_frag(const __bf16* lA, int mbase, int lane) {
    int m    = mbase + (lane & 15);
    int ksel = (lane >> 4) * 8;
    const __bf16* p = lA + m * KSTR + ksel;
    v16bf f;
#pragma unroll
    for (int j = 0; j < 8; ++j) { f[j] = p[j]; f[j + 8] = p[j + 16]; }
    return f;
}

// B fragment (32x16 bf16, K x N): lane l<16 holds col N=l with K=0..7,16..23;
// lane l>=16 holds col N=l-16 with K=8..15,24..31.  lB is [n][k], stride KSTR.
__device__ __forceinline__ v16bf load_b_frag(const __bf16* lB, int nbase, int lane) {
    int n    = nbase + (lane & 15);
    int koff = (lane >> 4) * 8;
    const __bf16* p = lB + n * KSTR + koff;
    v16bf f;
#pragma unroll
    for (int j = 0; j < 8; ++j) { f[j] = p[j]; f[j + 8] = p[j + 16]; }
    return f;
}

__device__ __forceinline__ v8f zero_v8f() {
    v8f z;
#pragma unroll
    for (int q = 0; q < 8; ++q) z[q] = 0.f;
    return z;
}

// --------------------------------- kernels ---------------------------------

__global__ void k_zero(float* __restrict__ out, int n, int* __restrict__ counts) {
    int i = blockIdx.x * blockDim.x + threadIdx.x;
    for (; i < n; i += gridDim.x * blockDim.x) out[i] = 0.f;
    if (blockIdx.x == 0 && threadIdx.x < E_) counts[threadIdx.x] = 0;
}

// Router: logits -> softmax -> top-8 -> renormalize. Also x fp32 -> bf16.
__global__ void k_router(const float* __restrict__ x, const float* __restrict__ wr,
                         int* __restrict__ out_ids, int* __restrict__ topk_id,
                         float* __restrict__ topk_w, int* __restrict__ counts,
                         __bf16* __restrict__ x_bf) {
    int t = blockIdx.x, tid = threadIdx.x;          // 64 threads: one expert each
    __shared__ float xs[H_];
    for (int i = tid; i < H_; i += 64) {
        float v = x[(size_t)t * H_ + i];
        xs[i] = v;
        x_bf[(size_t)t * H_ + i] = (__bf16)v;
    }
    __syncthreads();
    float acc = 0.f;
    for (int h = 0; h < H_; ++h) acc += xs[h] * wr[h * E_ + tid];
    __shared__ float lg[E_];
    lg[tid] = acc;
    __syncthreads();
    if (tid == 0) {
        float mx = lg[0];
        for (int e = 1; e < E_; ++e) mx = fmaxf(mx, lg[e]);
        float s = 0.f;
        for (int e = 0; e < E_; ++e) { float p = __expf(lg[e] - mx); lg[e] = p; s += p; }
        float inv = 1.f / s;
        for (int e = 0; e < E_; ++e) lg[e] *= inv;
        int ids[TOPK]; float wv[TOPK]; float wsum = 0.f;
        for (int k = 0; k < TOPK; ++k) {
            int bi = 0; float bv = -1.f;
            for (int e = 0; e < E_; ++e) if (lg[e] > bv) { bv = lg[e]; bi = e; }
            ids[k] = bi; wv[k] = bv; wsum += bv; lg[bi] = -2.f;
        }
        float invw = 1.f / wsum;
        for (int k = 0; k < TOPK; ++k) {
            int p = t * TOPK + k;
            topk_id[p] = ids[k];
            topk_w[p]  = wv[k] * invw;
            out_ids[p] = ids[k];
            atomicAdd(&counts[ids[k]], 1);
        }
    }
}

// Serial scan over 64 experts: 64-aligned segments + M-tile table.
__global__ void k_scan(const int* __restrict__ counts, int* __restrict__ seg_base,
                       int* __restrict__ fill, int* __restrict__ tile_e,
                       int* __restrict__ tile_m0, int* __restrict__ num_tiles) {
    if (blockIdx.x == 0 && threadIdx.x == 0) {
        int base = 0, nt = 0;
        for (int e = 0; e < E_; ++e) {
            seg_base[e] = base;
            fill[e] = 0;
            int c = counts[e];
            int ntile = (c + 63) >> 6;
            for (int i = 0; i < ntile; ++i) { tile_e[nt] = e; tile_m0[nt] = base + i * 64; ++nt; }
            base += ntile * 64;
        }
        *num_tiles = nt;
    }
}

// Zero padded rows in each expert segment; mark them unused.
__global__ void k_padzero(const int* __restrict__ counts, const int* __restrict__ seg_base,
                          int* __restrict__ row2tok, float* __restrict__ row_w,
                          __bf16* __restrict__ xs) {
    int e = blockIdx.x;
    int c = counts[e], b = seg_base[e];
    int cend = (c + 63) & ~63;
    for (int r = c; r < cend; ++r) {
        if (threadIdx.x == 0) { row2tok[b + r] = -1; row_w[b + r] = 0.f; }
        __bf16* row = xs + (size_t)(b + r) * H_;
        for (int i = threadIdx.x; i < H_; i += 256) row[i] = (__bf16)0.f;
    }
}

// Place each (token, expert) pair into its expert segment; gather bf16 row.
__global__ void k_gather(const int* __restrict__ topk_id, const float* __restrict__ topk_w,
                         const int* __restrict__ seg_base, int* __restrict__ fill,
                         int* __restrict__ row2tok, float* __restrict__ row_w,
                         const __bf16* __restrict__ x_bf, __bf16* __restrict__ xs) {
    int p = blockIdx.x;
    int t = p / TOPK;
    int e = topk_id[p];
    __shared__ int slot_s;
    if (threadIdx.x == 0) slot_s = seg_base[e] + atomicAdd(&fill[e], 1);
    __syncthreads();
    int slot = slot_s;
    if (threadIdx.x == 0) { row2tok[slot] = t; row_w[slot] = topk_w[p]; }
    const uint4* src = (const uint4*)(x_bf + (size_t)t * H_);
    uint4* dst = (uint4*)(xs + (size_t)slot * H_);
    for (int i = threadIdx.x; i < H_ * 2 / 16; i += 64) dst[i] = src[i];
}

// GEMM1: Hout = silu(A @ Wg) * (A @ Wu), A bf16 [rows,H_], W fp32 [H_,Ntot].
// Macro tile 64(M) x 128(N), 8 waves (2x4), wave = 32x32 = 2x2 WMMA subtiles.
// Double-buffered LDS; A staged via async global->LDS, B converted via VGPRs.
template <bool EXPERT>
__global__ __launch_bounds__(256) void k_gemm_gateup(
    const __bf16* __restrict__ A, const float* __restrict__ Wg, const float* __restrict__ Wu,
    __bf16* __restrict__ Hout, int Ntot,
    const int* __restrict__ tile_e, const int* __restrict__ tile_m0,
    const int* __restrict__ num_tiles) {
    int tile = blockIdx.x;
    size_t wbase = 0;
    int m0;
    if constexpr (EXPERT) {
        if (tile >= *num_tiles) return;
        int e = tile_e[tile];
        m0 = tile_m0[tile];
        wbase = (size_t)e * (size_t)H_ * (size_t)Ntot;
    } else {
        m0 = tile * 64;
    }
    int n0 = blockIdx.y * 128;

    __shared__ __bf16 lA[2][64 * KSTR];
    __shared__ __bf16 lBg[2][128 * KSTR];
    __shared__ __bf16 lBu[2][128 * KSTR];

    int tid = threadIdx.x, lane = tid & 31, wv = tid >> 5;
    int wm = (wv >> 2) * 32, wn = (wv & 3) * 32;

    // per-thread tile-copy coordinates
    int ar = tid >> 2, ac = (tid & 3) * 8;            // A: one 16B chunk / thread

    v8f accg[2][2], accu[2][2];
#pragma unroll
    for (int a = 0; a < 2; ++a)
#pragma unroll
        for (int b = 0; b < 2; ++b) { accg[a][b] = zero_v8f(); accu[a][b] = zero_v8f(); }

    // ---- prologue: stage tile kk=0 into buffer 0 ----
    async_copy16(&A[(size_t)(m0 + ar) * H_ + ac], &lA[0][ar * KSTR + ac]);
#pragma unroll
    for (int j = 0; j < 16; ++j) {
        int idx = tid + j * 256;
        int k = idx >> 7, n = idx & 127;
        size_t gi = wbase + (size_t)k * Ntot + (n0 + n);
        lBg[0][n * KSTR + k] = (__bf16)Wg[gi];
        lBu[0][n * KSTR + k] = (__bf16)Wu[gi];
    }
    wait_async();
    __syncthreads();

    for (int kk = 0; kk < H_; kk += 32) {
        int cur = (kk >> 5) & 1, nxt = cur ^ 1;
        int kk2 = kk + 32;
        float bg[16], bu[16];
        if (kk2 < H_) {
            async_copy16(&A[(size_t)(m0 + ar) * H_ + kk2 + ac], &lA[nxt][ar * KSTR + ac]);
#pragma unroll
            for (int j = 0; j < 16; ++j) {
                int idx = tid + j * 256;
                int k = idx >> 7, n = idx & 127;
                size_t gi = wbase + (size_t)(kk2 + k) * Ntot + (n0 + n);
                bg[j] = Wg[gi];
                bu[j] = Wu[gi];
            }
        }

        v16bf af[2], bgf[2], buf[2];
#pragma unroll
        for (int s = 0; s < 2; ++s) {
            af[s]  = load_a_frag(&lA[cur][0], wm + s * 16, lane);
            bgf[s] = load_b_frag(&lBg[cur][0], wn + s * 16, lane);
            buf[s] = load_b_frag(&lBu[cur][0], wn + s * 16, lane);
        }
#pragma unroll
        for (int sm = 0; sm < 2; ++sm)
#pragma unroll
            for (int sn = 0; sn < 2; ++sn) {
                accg[sm][sn] = wmma_bf16(af[sm], bgf[sn], accg[sm][sn]);
                accu[sm][sn] = wmma_bf16(af[sm], buf[sn], accu[sm][sn]);
            }

        if (kk2 < H_) {
#pragma unroll
            for (int j = 0; j < 16; ++j) {
                int idx = tid + j * 256;
                int k = idx >> 7, n = idx & 127;
                lBg[nxt][n * KSTR + k] = (__bf16)bg[j];
                lBu[nxt][n * KSTR + k] = (__bf16)bu[j];
            }
        }
        wait_async();
        __syncthreads();
    }

    int lhalf = lane >> 4, lcol = lane & 15;
#pragma unroll
    for (int sm = 0; sm < 2; ++sm)
#pragma unroll
        for (int sn = 0; sn < 2; ++sn)
#pragma unroll
            for (int r = 0; r < 8; ++r) {
                int m = m0 + wm + sm * 16 + lhalf * 8 + r;
                int n = n0 + wn + sn * 16 + lcol;
                float g = accg[sm][sn][r], u = accu[sm][sn][r];
                Hout[(size_t)m * Ntot + n] = (__bf16)(fast_silu(g) * u);
            }
}

// GEMM2: Out[dst] += scale(row) * (A @ Wd), A bf16 [rows,Kdim], Wd fp32 [Kdim,H_].
template <bool EXPERT>
__global__ __launch_bounds__(256) void k_gemm_down(
    const __bf16* __restrict__ A, const float* __restrict__ Wd, int Kdim,
    float* __restrict__ Out,
    const int* __restrict__ tile_e, const int* __restrict__ tile_m0,
    const int* __restrict__ num_tiles,
    const int* __restrict__ row2tok, const float* __restrict__ row_w,
    const float* __restrict__ gate_sig) {
    int tile = blockIdx.x;
    size_t wbase = 0;
    int m0;
    if constexpr (EXPERT) {
        if (tile >= *num_tiles) return;
        int e = tile_e[tile];
        m0 = tile_m0[tile];
        wbase = (size_t)e * (size_t)Kdim * (size_t)H_;
    } else {
        m0 = tile * 64;
    }
    int n0 = blockIdx.y * 128;

    __shared__ __bf16 lA[2][64 * KSTR];
    __shared__ __bf16 lB[2][128 * KSTR];

    int tid = threadIdx.x, lane = tid & 31, wv = tid >> 5;
    int wm = (wv >> 2) * 32, wn = (wv & 3) * 32;
    int ar = tid >> 2, ac = (tid & 3) * 8;

    v8f acc[2][2];
#pragma unroll
    for (int a = 0; a < 2; ++a)
#pragma unroll
        for (int b = 0; b < 2; ++b) acc[a][b] = zero_v8f();

    // ---- prologue ----
    async_copy16(&A[(size_t)(m0 + ar) * Kdim + ac], &lA[0][ar * KSTR + ac]);
#pragma unroll
    for (int j = 0; j < 16; ++j) {
        int idx = tid + j * 256;
        int k = idx >> 7, n = idx & 127;
        lB[0][n * KSTR + k] = (__bf16)Wd[wbase + (size_t)k * H_ + (n0 + n)];
    }
    wait_async();
    __syncthreads();

    for (int kk = 0; kk < Kdim; kk += 32) {
        int cur = (kk >> 5) & 1, nxt = cur ^ 1;
        int kk2 = kk + 32;
        float bd[16];
        if (kk2 < Kdim) {
            async_copy16(&A[(size_t)(m0 + ar) * Kdim + kk2 + ac], &lA[nxt][ar * KSTR + ac]);
#pragma unroll
            for (int j = 0; j < 16; ++j) {
                int idx = tid + j * 256;
                int k = idx >> 7, n = idx & 127;
                bd[j] = Wd[wbase + (size_t)(kk2 + k) * H_ + (n0 + n)];
            }
        }

        v16bf af[2], bf[2];
#pragma unroll
        for (int s = 0; s < 2; ++s) {
            af[s] = load_a_frag(&lA[cur][0], wm + s * 16, lane);
            bf[s] = load_b_frag(&lB[cur][0], wn + s * 16, lane);
        }
#pragma unroll
        for (int sm = 0; sm < 2; ++sm)
#pragma unroll
            for (int sn = 0; sn < 2; ++sn)
                acc[sm][sn] = wmma_bf16(af[sm], bf[sn], acc[sm][sn]);

        if (kk2 < Kdim) {
#pragma unroll
            for (int j = 0; j < 16; ++j) {
                int idx = tid + j * 256;
                int k = idx >> 7, n = idx & 127;
                lB[nxt][n * KSTR + k] = (__bf16)bd[j];
            }
        }
        wait_async();
        __syncthreads();
    }

    int lhalf = lane >> 4, lcol = lane & 15;
#pragma unroll
    for (int sm = 0; sm < 2; ++sm)
#pragma unroll
        for (int sn = 0; sn < 2; ++sn)
#pragma unroll
            for (int r = 0; r < 8; ++r) {
                int m = m0 + wm + sm * 16 + lhalf * 8 + r;
                int n = n0 + wn + sn * 16 + lcol;
                float v = acc[sm][sn][r];
                int dst;
                float s;
                if constexpr (EXPERT) {
                    dst = row2tok[m];
                    if (dst < 0) continue;
                    s = row_w[m];
                } else {
                    dst = m;
                    s = gate_sig[m];
                }
                atomicAdd(&Out[(size_t)dst * H_ + n], v * s);
            }
}

// Shared-expert per-token sigmoid gate: sigmoid(x . w_shared_gate). 1 wave/token.
__global__ void k_sgate(const float* __restrict__ x, const float* __restrict__ wg,
                        float* __restrict__ gate_sig) {
    int wv = threadIdx.x >> 5, lane = threadIdx.x & 31;
    int t = blockIdx.x * 8 + wv;
    const float* xr = x + (size_t)t * H_;
    float acc = 0.f;
    for (int h = lane; h < H_; h += 32) acc += xr[h] * wg[h];
#pragma unroll
    for (int off = 16; off; off >>= 1) acc += __shfl_xor(acc, off, 32);
    if (lane == 0) gate_sig[t] = fast_sigmoid(acc);
}

// --------------------------------- launch ----------------------------------

extern "C" void kernel_launch(void* const* d_in, const int* in_sizes, int n_in,
                              void* d_out, int out_size, void* d_ws, size_t ws_size,
                              hipStream_t stream) {
    const float* x        = (const float*)d_in[0];   // [T,H]
    const float* w_router = (const float*)d_in[1];   // [H,E]
    const float* w_gate_e = (const float*)d_in[2];   // [E,H,I]
    const float* w_up_e   = (const float*)d_in[3];   // [E,H,I]
    const float* w_down_e = (const float*)d_in[4];   // [E,I,H]
    const float* ws_gate  = (const float*)d_in[5];   // [H,IS]
    const float* ws_up    = (const float*)d_in[6];   // [H,IS]
    const float* ws_down  = (const float*)d_in[7];   // [IS,H]
    const float* w_sgate  = (const float*)d_in[8];   // [H,1]

    float* out    = (float*)d_out;                   // [T,H] fp32
    int*   outids = (int*)(out + (size_t)T_ * H_);   // [T,K] int32 (bit-stored)

    // workspace carve-up (256B aligned)
    char* base = (char*)d_ws;
    size_t off = 0;
    auto carve = [&](size_t bytes) -> void* {
        void* r = base + off;
        off = (off + bytes + 255) & ~(size_t)255;
        return r;
    };
    __bf16* x_bf    = (__bf16*)carve((size_t)T_ * H_ * 2);
    __bf16* xs      = (__bf16*)carve((size_t)CAP * H_ * 2);
    __bf16* hexp    = (__bf16*)carve((size_t)CAP * I_ * 2);
    __bf16* hsh     = (__bf16*)carve((size_t)T_ * IS_ * 2);
    int*    topk_id = (int*)carve((size_t)T_ * TOPK * 4);
    float*  topk_w  = (float*)carve((size_t)T_ * TOPK * 4);
    int*    row2tok = (int*)carve((size_t)CAP * 4);
    float*  row_w   = (float*)carve((size_t)CAP * 4);
    float*  gatesig = (float*)carve((size_t)T_ * 4);
    int*    counts  = (int*)carve(E_ * 4);
    int*    segb    = (int*)carve(E_ * 4);
    int*    fill    = (int*)carve(E_ * 4);
    int*    tile_e  = (int*)carve(MAXT * 4);
    int*    tile_m0 = (int*)carve(MAXT * 4);
    int*    ntiles  = (int*)carve(4);

    // 1) zero routed output + counters
    k_zero<<<4096, 256, 0, stream>>>(out, T_ * H_, counts);

    // 2) router + x->bf16
    k_router<<<T_, 64, 0, stream>>>(x, w_router, outids, topk_id, topk_w, counts, x_bf);

    // 3) segments + tile table
    k_scan<<<1, 1, 0, stream>>>(counts, segb, fill, tile_e, tile_m0, ntiles);

    // 4) zero pad rows, then gather dispatched activations
    k_padzero<<<E_, 256, 0, stream>>>(counts, segb, row2tok, row_w, xs);
    k_gather<<<T_ * TOPK, 64, 0, stream>>>(topk_id, topk_w, segb, fill, row2tok, row_w, x_bf, xs);

    // 5) expert FFN: gate+up fused (K=H_, N=I_), then down + scatter (K=I_, N=H_)
    {
        dim3 g1(MAXT, I_ / 128);
        k_gemm_gateup<true><<<g1, 256, 0, stream>>>(xs, w_gate_e, w_up_e, hexp, I_,
                                                    tile_e, tile_m0, ntiles);
        dim3 g2(MAXT, H_ / 128);
        k_gemm_down<true><<<g2, 256, 0, stream>>>(hexp, w_down_e, I_, out,
                                                  tile_e, tile_m0, ntiles,
                                                  row2tok, row_w, nullptr);
    }

    // 6) shared expert: sigmoid gate, gate+up (N=IS_), down scaled by gate (K=IS_)
    k_sgate<<<T_ / 8, 256, 0, stream>>>(x, w_sgate, gatesig);
    {
        dim3 g1(T_ / 64, IS_ / 128);
        k_gemm_gateup<false><<<g1, 256, 0, stream>>>(x_bf, ws_gate, ws_up, hsh, IS_,
                                                     nullptr, nullptr, nullptr);
        dim3 g2(T_ / 64, H_ / 128);
        k_gemm_down<false><<<g2, 256, 0, stream>>>(hsh, ws_down, IS_, out,
                                                   nullptr, nullptr, nullptr,
                                                   nullptr, nullptr, gatesig);
    }

    (void)in_sizes; (void)n_in; (void)out_size; (void)ws_size;
}
